// SPINN_24670292148981
// MI455X (gfx1250) — matrix operand order repaired and checked by reading, not compiled, not run on hardware
//
#include <hip/hip_runtime.h>
#include <hip/hip_bf16.h>

// ---------------------------------------------------------------------------
// SPINN forward for MI455X (gfx1250, wave32, WMMA + TDM).
// Shapes: B=256, T=256, WD=128, H=128, TD=64, NT=2T-3=509.
// ---------------------------------------------------------------------------

#define BSZ   256
#define TSZ   256
#define WD    128
#define HN    128
#define H2    256      // 2H
#define TDIM  64
#define NTR   509      // 2T-3
#define RB    16       // batch rows per scan block

typedef __attribute__((ext_vector_type(16))) __bf16 v16bf;
typedef __attribute__((ext_vector_type(8)))  float  v8f;
typedef __attribute__((ext_vector_type(4)))  unsigned int v4u;
typedef __attribute__((ext_vector_type(8)))  int v8i;
typedef __attribute__((ext_vector_type(4)))  int v4i;

// ---- WMMA fragment helpers (contiguous k = 16h + s in both A and B, wave32) ---------------
// A (16x32 bf16): lanes 0-15 -> M=lane, K = kb..kb+15
//                 lanes 16-31 -> M=lane-16, K = kb+16..kb+31
__device__ inline v16bf lds_loadA(const __bf16* s, int stride, int kb, int lane) {
  const __bf16* row = s + (lane & 15) * stride + kb + ((lane < 16) ? 0 : 16);
  v16bf a;
#pragma unroll
  for (int i = 0; i < 8; ++i) { a[i] = row[i]; a[8 + i] = row[8 + i]; }
  return a;
}

// B (32x16 bf16) from row-major weight W[N][K]: lane holds N = n0 + lane%16,
// K = kb + (lane<16 ? 0 : 16) + i  -> 16 contiguous bf16 (32B) per lane.
__device__ inline v16bf loadB(const __bf16* W, int kdim, int n0, int kb, int lane) {
  const __bf16* p = W + (size_t)(n0 + (lane & 15)) * kdim + kb + ((lane < 16) ? 0 : 16);
  v16bf b;
#pragma unroll
  for (int i = 0; i < 16; ++i) b[i] = p[i];
  return b;
}

__device__ inline float sigm(float x) { return 1.0f / (1.0f + __expf(-x)); }

// ---- Tensor Data Mover: 2-D bf16 tensor -> LDS (D# per ISA ch.8) ----------
// k = row length (elements), n = #rows, stride = row stride (elements).
// This toolchain exposes the 6-arg builtin:
//   (v4u group0, v8i group1, v4i group2, v4i group3, v8i extra, i32 cpol)
__device__ inline void tdm_load_2d_bf16(unsigned lds_addr, const void* gptr,
                                        unsigned k, unsigned n, unsigned stride) {
#if __has_builtin(__builtin_amdgcn_tensor_load_to_lds)
  const unsigned long long ga = (unsigned long long)(uintptr_t)gptr;
  v4u g0;
  g0[0] = 1u;                                        // count=1 (valid user D#)
  g0[1] = lds_addr;                                  // lds_addr (bytes)
  g0[2] = (unsigned)(ga & 0xFFFFFFFFu);              // global_addr[31:0]
  g0[3] = (unsigned)((ga >> 32) & 0x01FFFFFFu) | (2u << 30); // [56:32] | type=2
  v8i g1;
  g1[0] = (int)(1u << 16);                           // data_size=1 -> 2 bytes
  g1[1] = (int)((k & 0xFFFFu) << 16);                // tensor_dim0[15:0] @63:48
  g1[2] = (int)(((k >> 16) & 0xFFFFu) | ((n & 0xFFFFu) << 16)); // dim0 hi | dim1 lo
  g1[3] = (int)(((n >> 16) & 0xFFFFu) | ((k & 0xFFFFu) << 16)); // dim1 hi | tile_dim0
  g1[4] = (int)(n & 0xFFFFu);                        // tile_dim1 (tile_dim2=0)
  g1[5] = (int)stride;                               // tensor_dim0_stride[31:0]
  g1[6] = 0;
  g1[7] = 0;
  const v4i z4 = {0, 0, 0, 0};
  const v8i z8 = {0, 0, 0, 0, 0, 0, 0, 0};
  __builtin_amdgcn_tensor_load_to_lds(g0, g1, z4, z4, z8, 0);
#else
  (void)lds_addr; (void)gptr; (void)k; (void)n; (void)stride;
#endif
}

// ---------------------------------------------------------------------------
// K0: convert the 509x-reused weights to bf16 (same row-major [N][K] layout).
// ---------------------------------------------------------------------------
__global__ __launch_bounds__(256) void wconvert(const float* __restrict__ Wih,
                                                const float* __restrict__ Whh,
                                                const float* __restrict__ Wr,
                                                __bf16* dWih, __bf16* dWhh, __bf16* dWr) {
  int i = blockIdx.x * 256 + threadIdx.x;
  if (i < 256 * 384) dWih[i] = (__bf16)Wih[i];
  if (i < 256 * 64)  dWhh[i] = (__bf16)Whh[i];
  if (i < 640 * 320) dWr[i]  = (__bf16)Wr[i];
}

// ---------------------------------------------------------------------------
// K1: buf = tokens @ Ww^T + bw   (65536 x 256, K=128). One 16x16 tile / wave.
// ---------------------------------------------------------------------------
__global__ __launch_bounds__(256) void proj_gemm(const float* __restrict__ tokens,
                                                 const float* __restrict__ Ww,
                                                 const float* __restrict__ bw,
                                                 float* __restrict__ buf) {
  const int lane = threadIdx.x & 31;
  const int wave = threadIdx.x >> 5;
  const int tile = blockIdx.x * 8 + wave;    // 65536 tiles total
  const int mt = tile >> 4;                  // 0..4095 (M tiles of 16 rows)
  const int nt = tile & 15;                  // 0..15   (N tiles of 16 cols)

  const float* arow = tokens + (size_t)(mt * 16 + (lane & 15)) * WD;
  const int koffA = (lane < 16) ? 0 : 16;
  const int koffB = (lane < 16) ? 0 : 16;
  const int n = nt * 16 + (lane & 15);
  const float* brow = Ww + (size_t)n * WD;

  v8f c = {};
#pragma unroll
  for (int ks = 0; ks < 4; ++ks) {
    const int kb = ks * 32;
    v16bf a, b;
#pragma unroll
    for (int i = 0; i < 8; ++i) {
      a[i]     = (__bf16)arow[kb + koffA + i];
      a[8 + i] = (__bf16)arow[kb + 8 + koffA + i];
    }
#pragma unroll
    for (int i = 0; i < 16; ++i) b[i] = (__bf16)brow[kb + koffB + i];
    c = __builtin_amdgcn_wmma_f32_16x16x32_bf16(false, a, false, b, (short)0, c, false, false);
  }
  const float bias = bw[n];
  const int moff = (lane < 16) ? 0 : 8;
#pragma unroll
  for (int v = 0; v < 8; ++v)
    buf[(size_t)(mt * 16 + v + moff) * H2 + n] = c[v] + bias;
}

// ---------------------------------------------------------------------------
// K2: persistent shift-reduce scan. 16 blocks x 256 threads (8 waves).
// Block b owns batch rows [16b, 16b+16): full 509-step recurrence, no global
// sync. Stack + buf live in L2 (135 MB < 192 MB). Tracker weights are staged
// into LDS once via the Tensor Data Mover.
// ---------------------------------------------------------------------------
__global__ __launch_bounds__(256) void spinn_scan(
    const float* __restrict__ buf, float* __restrict__ stack,
    const __bf16* __restrict__ gWih, const __bf16* __restrict__ gWhh,
    const __bf16* __restrict__ Wr,
    const float* __restrict__ bih, const float* __restrict__ bhh,
    const float* __restrict__ br,
    const float* __restrict__ Wt, const float* __restrict__ bt,
    const int* __restrict__ trans, float* __restrict__ out) {
  __shared__ __bf16 sWih[256 * 384];  // 196,608 B  (TDM-staged, reused 509x)
  __shared__ __bf16 sWhh[256 * 64];   //  32,768 B  (TDM-staged)
  __shared__ __bf16 xA[RB * 384];     // [buf_h | left_h | right_h] bf16
  __shared__ __bf16 thA[RB * 64];     // tracker h, bf16 (A for Whh GEMM)
  __shared__ __bf16 lrtA[RB * 320];   // [left_h | right_h | th] bf16
  __shared__ float  sGates[RB * 256]; // tracker gate pre-activations
  __shared__ float  sRout[RB * 640];  // reducer pre-activations
  __shared__ float  sTh[RB * 64];
  __shared__ float  sTc[RB * 64];
  __shared__ int sPtr[RB], sBptr[RB], sLp[RB], sRp[RB], sBq[RB], sPos[RB], sShift[RB];

  const int tid  = threadIdx.x;
  const int lane = tid & 31;
  const int wave = tid >> 5;
  const int gr0  = blockIdx.x * RB;

  // ---- stage tracker weights into LDS with the Tensor Data Mover ----------
#if __has_builtin(__builtin_amdgcn_tensor_load_to_lds)
  if (tid == 0) {
    tdm_load_2d_bf16((unsigned)(uintptr_t)(void*)sWih, gWih, 384u, 256u, 384u);
    tdm_load_2d_bf16((unsigned)(uintptr_t)(void*)sWhh, gWhh, 64u, 256u, 64u);
    __builtin_amdgcn_s_wait_tensorcnt(0);
  }
#else
  for (int i = tid; i < 256 * 384; i += 256) sWih[i] = gWih[i];
  for (int i = tid; i < 256 * 64;  i += 256) sWhh[i] = gWhh[i];
#endif

  // ---- init: zero stack slots 0,1 for our rows; zero tracker state --------
  for (int idx = tid; idx < RB * 2 * H2; idx += 256) {
    const int r = idx >> 9, slot = (idx >> 8) & 1, c = idx & 255;
    stack[((size_t)(gr0 + r) * (TSZ + 2) + slot) * H2 + c] = 0.0f;
  }
  for (int idx = tid; idx < RB * TDIM; idx += 256) {
    sTh[idx] = 0.0f; sTc[idx] = 0.0f; thA[idx] = (__bf16)0.0f;
  }
  if (tid < RB) { sPtr[tid] = 2; sBptr[tid] = 0; }
  __syncthreads();

  for (int stp = 0; stp < NTR; ++stp) {
    // ---- phase A: decode transition, snapshot pointers, advance them ------
    if (tid < RB) {
      const int r = tid;
      const int ptr = sPtr[r], bp = sBptr[r];
      const int tr = trans[(size_t)(gr0 + r) * NTR + stp];
      const int sh = (tr == 0);
      sShift[r] = sh;
      sLp[r] = ptr - 2; sRp[r] = ptr - 1; sBq[r] = bp;
      sPos[r]  = sh ? ptr : ptr - 2;
      sPtr[r]  = sh ? ptr + 1 : ptr - 1;
      sBptr[r] = bp + sh;
    }
    __syncthreads();

    // ---- gather h-halves of stack top-2 + buffer head -> LDS bf16 ---------
    for (int idx = tid; idx < RB * HN; idx += 256) {
      const int r = idx >> 7, c = idx & 127;
      const size_t rowb = (size_t)(gr0 + r) * (TSZ + 2);
      const float vL = stack[(rowb + sLp[r]) * H2 + c];
      const float vR = stack[(rowb + sRp[r]) * H2 + c];
      const float vB = buf[((size_t)(gr0 + r) * TSZ + sBq[r]) * H2 + c];
      xA[r * 384 + c]         = (__bf16)vB;
      xA[r * 384 + 128 + c]   = (__bf16)vL;
      xA[r * 384 + 256 + c]   = (__bf16)vR;
      lrtA[r * 320 + c]       = (__bf16)vL;
      lrtA[r * 320 + 128 + c] = (__bf16)vR;
    }
    __syncthreads();

    // ---- tracker gates: (16x384)@Wih^T + (16x64)@Whh^T -> 16x256 ----------
    for (int t = wave; t < 16; t += 8) {
      v8f c = {};
      const int n0 = t * 16;
#pragma unroll
      for (int ks = 0; ks < 12; ++ks) {
        v16bf a = lds_loadA(xA, 384, ks * 32, lane);
        v16bf b = loadB(sWih, 384, n0, ks * 32, lane);
        c = __builtin_amdgcn_wmma_f32_16x16x32_bf16(false, a, false, b, (short)0, c, false, false);
      }
#pragma unroll
      for (int ks = 0; ks < 2; ++ks) {
        v16bf a = lds_loadA(thA, 64, ks * 32, lane);
        v16bf b = loadB(sWhh, 64, n0, ks * 32, lane);
        c = __builtin_amdgcn_wmma_f32_16x16x32_bf16(false, a, false, b, (short)0, c, false, false);
      }
      const int n = n0 + (lane & 15);
      const float bsum = bih[n] + bhh[n];
      const int moff = (lane < 16) ? 0 : 8;
#pragma unroll
      for (int v = 0; v < 8; ++v) sGates[(v + moff) * 256 + n] = c[v] + bsum;
    }
    __syncthreads();

    // ---- tracker LSTM elementwise (torch gate order i,f,g,o) --------------
    for (int idx = tid; idx < RB * TDIM; idx += 256) {
      const int r = idx >> 6, j = idx & 63;
      const float* g = &sGates[r * 256];
      const float gi = g[j], gf = g[64 + j], gg = g[128 + j], go = g[192 + j];
      const float tc = sigm(gf) * sTc[idx] + sigm(gi) * tanhf(gg);
      const float th = sigm(go) * tanhf(tc);
      sTc[idx] = tc; sTh[idx] = th;
      thA[r * 64 + j]         = (__bf16)th;
      lrtA[r * 320 + 256 + j] = (__bf16)th;
    }
    __syncthreads();

    // ---- reducer: (16x320)@Wr^T -> 16x640 (Wr streamed from L2) -----------
    for (int t = wave; t < 40; t += 8) {
      v8f c = {};
      const int n0 = t * 16;
#pragma unroll
      for (int ks = 0; ks < 10; ++ks) {
        v16bf a = lds_loadA(lrtA, 320, ks * 32, lane);
        v16bf b = loadB(Wr, 320, n0, ks * 32, lane);
        c = __builtin_amdgcn_wmma_f32_16x16x32_bf16(false, a, false, b, (short)0, c, false, false);
      }
      const int n = n0 + (lane & 15);
      const float bb = br[n];
      const int moff = (lane < 16) ? 0 : 8;
#pragma unroll
      for (int v = 0; v < 8; ++v) sRout[(v + moff) * 640 + n] = c[v] + bb;
    }
    __syncthreads();

    // ---- logits (tiny 64x2 matvec per row) --------------------------------
    if (tid < RB) {
      const int r = tid;
      float l0 = bt[0], l1 = bt[1];
#pragma unroll 4
      for (int j = 0; j < TDIM; ++j) {
        const float th = sTh[r * 64 + j];
        l0 += th * Wt[j];
        l1 += th * Wt[64 + j];
      }
      const size_t o = (size_t)BSZ * HN + (size_t)(gr0 + r) * NTR * 2 + (size_t)stp * 2;
      out[o] = l0; out[o + 1] = l1;
    }

    // ---- parent compose + stack write (shift pushes buf_top) --------------
    // c-halves of left/right are re-read from L2-resident stack lines; for a
    // reduce row the write target (sPos == sLp) aliases only the element this
    // same thread read, so read-then-write per (r,j) is race-free.
    for (int idx = tid; idx < RB * HN; idx += 256) {
      const int r = idx >> 7, j = idx & 127;
      const float* q = &sRout[r * 640];
      const float ri = q[j], rfl = q[128 + j], rfr = q[256 + j];
      const float ro = q[384 + j], rg = q[512 + j];
      const size_t rowb = (size_t)(gr0 + r) * (TSZ + 2);
      const float lc = stack[(rowb + sLp[r]) * H2 + 128 + j];
      const float rc = stack[(rowb + sRp[r]) * H2 + 128 + j];
      const float pc = sigm(rfl) * lc + sigm(rfr) * rc + sigm(ri) * tanhf(rg);
      const float ph = sigm(ro) * tanhf(pc);
      float* dst = &stack[(rowb + sPos[r]) * H2];
      if (sShift[r]) {
        const float* bsrc = &buf[((size_t)(gr0 + r) * TSZ + sBq[r]) * H2];
        dst[j] = bsrc[j]; dst[128 + j] = bsrc[128 + j];
      } else {
        dst[j] = ph; dst[128 + j] = pc;
      }
    }
    __syncthreads();
  }

  // ---- root_h = stack[ptr-1][:H] -------------------------------------------
  for (int idx = tid; idx < RB * HN; idx += 256) {
    const int r = idx >> 7, j = idx & 127;
    const size_t rowb = (size_t)(gr0 + r) * (TSZ + 2);
    out[(size_t)(gr0 + r) * HN + j] = stack[(rowb + sPtr[r] - 1) * H2 + j];
  }
}

// ---------------------------------------------------------------------------
extern "C" void kernel_launch(void* const* d_in, const int* in_sizes, int n_in,
                              void* d_out, int out_size, void* d_ws, size_t ws_size,
                              hipStream_t stream) {
  const float* tokens = (const float*)d_in[0];
  const float* Ww     = (const float*)d_in[1];
  const float* bw     = (const float*)d_in[2];
  const float* Wr     = (const float*)d_in[3];
  const float* br     = (const float*)d_in[4];
  const float* Wih    = (const float*)d_in[5];
  const float* Whh    = (const float*)d_in[6];
  const float* bih    = (const float*)d_in[7];
  const float* bhh    = (const float*)d_in[8];
  const float* Wt     = (const float*)d_in[9];
  const float* bt     = (const float*)d_in[10];
  const int*   trans  = (const int*)d_in[11];
  float* out = (float*)d_out;

  char* ws = (char*)d_ws;
  float*  buf   = (float*)(ws + 0);            //  67,108,864 B: (B*T, 2H) fp32
  float*  stack = (float*)(ws + 67108864);     //  67,633,152 B: (B, T+2, 2H) fp32
  __bf16* dWih  = (__bf16*)(ws + 134742016);   //     196,608 B
  __bf16* dWhh  = (__bf16*)(ws + 134938624);   //      32,768 B
  __bf16* dWr   = (__bf16*)(ws + 134971392);   //     409,600 B

  wconvert<<<800, 256, 0, stream>>>(Wih, Whh, Wr, dWih, dWhh, dWr);
  proj_gemm<<<8192, 256, 0, stream>>>(tokens, Ww, bw, buf);
  spinn_scan<<<BSZ / RB, 256, 0, stream>>>(buf, stack, dWih, dWhh, dWr,
                                           bih, bhh, br, Wt, bt, trans, out);
}
